// SimABlock_66451734004194
// MI455X (gfx1250) — matrix operand order repaired and checked by reading, not compiled, not run on hardware
//
#include <hip/hip_runtime.h>

// ---------------------------------------------------------------------------
// SimA transformer block for MI455X (gfx1250): WMMA f16 GEMMs + fp32 epilogues
// GEMM staging uses CDNA5 async global->LDS DMA (ASYNCcnt) with double buffer.
// ---------------------------------------------------------------------------

typedef __attribute__((ext_vector_type(16))) _Float16 v16h;
typedef __attribute__((ext_vector_type(8)))  float    v8f;

#define TILE_M 128
#define TILE_N 128
#define TILE_K 32

__device__ __forceinline__ float gelu_exact(float x) {
    return 0.5f * x * (1.0f + erff(x * 0.70710678118654752f));
}

// ---------------------------------------------------------------------------
// f32 -> f16 conversion (weights)
// ---------------------------------------------------------------------------
__global__ __launch_bounds__(256) void f32_to_f16_kernel(
    const float* __restrict__ src, _Float16* __restrict__ dst, int n) {
    int i = blockIdx.x * 256 + threadIdx.x;
    if (i < n) dst[i] = (_Float16)src[i];
}

// ---------------------------------------------------------------------------
// LayerNorm over C=512, output f16 row-major [M, C]
// ---------------------------------------------------------------------------
__global__ __launch_bounds__(256) void ln_f16_kernel(
    const float* __restrict__ X, const float* __restrict__ g,
    const float* __restrict__ b, _Float16* __restrict__ out, int C) {
    __shared__ float r0[256];
    __shared__ float r1[256];
    const int tid = threadIdx.x;
    const size_t row = blockIdx.x;
    const float* xr = X + row * C;
    float x0 = xr[tid], x1 = xr[tid + 256];
    r0[tid] = x0 + x1;
    r1[tid] = x0 * x0 + x1 * x1;
    __syncthreads();
    for (int off = 128; off > 0; off >>= 1) {
        if (tid < off) { r0[tid] += r0[tid + off]; r1[tid] += r1[tid + off]; }
        __syncthreads();
    }
    const float mean = r0[0] * (1.0f / 512.0f);
    const float var  = r1[0] * (1.0f / 512.0f) - mean * mean;
    const float inv  = rsqrtf(var + 1e-5f);
    out[row * C + tid]       = (_Float16)((x0 - mean) * inv * g[tid] + b[tid]);
    out[row * C + tid + 256] = (_Float16)((x1 - mean) * inv * g[tid + 256] + b[tid + 256]);
}

// LayerNorm, output f32 transposed to [B, C, N] (for depthwise conv stage)
__global__ __launch_bounds__(256) void ln_f32_tr_kernel(
    const float* __restrict__ X, const float* __restrict__ g,
    const float* __restrict__ b, float* __restrict__ out, int C, int N) {
    __shared__ float r0[256];
    __shared__ float r1[256];
    const int tid = threadIdx.x;
    const int row = blockIdx.x;          // row = bi*N + n
    const int bi = row / N;
    const int n  = row - bi * N;
    const float* xr = X + (size_t)row * C;
    float x0 = xr[tid], x1 = xr[tid + 256];
    r0[tid] = x0 + x1;
    r1[tid] = x0 * x0 + x1 * x1;
    __syncthreads();
    for (int off = 128; off > 0; off >>= 1) {
        if (tid < off) { r0[tid] += r0[tid + off]; r1[tid] += r1[tid + off]; }
        __syncthreads();
    }
    const float mean = r0[0] * (1.0f / 512.0f);
    const float var  = r1[0] * (1.0f / 512.0f) - mean * mean;
    const float inv  = rsqrtf(var + 1e-5f);
    float* ob = out + ((size_t)bi * C) * N + n;
    ob[(size_t)tid * N]         = (x0 - mean) * inv * g[tid] + b[tid];
    ob[(size_t)(tid + 256) * N] = (x1 - mean) * inv * g[tid + 256] + b[tid + 256];
}

// ---------------------------------------------------------------------------
// Tiled WMMA GEMM: Out = epilogue( A16[M,K] @ W16[K,Nn] )
//   block tile 128x128x32, 8 waves, each wave 64 (M) x 32 (N)
//   Tiles staged with global_load_async_to_lds_b128 (ASYNCcnt), double buffer.
//   EPI 0: out f16 (qkv)            1: outF = resid + gamma*(acc+bias)  (proj)
//   EPI 2: out f16 = gelu(acc+bias) 3: outF += gamma*(acc+bias)         (fc2)
// ---------------------------------------------------------------------------
template <int EPI>
__global__ __launch_bounds__(256) void gemm_wmma_kernel(
    const _Float16* __restrict__ A, const _Float16* __restrict__ B,
    int M, int K, int Nn,
    const float* __restrict__ bias, const float* __restrict__ gamma,
    const float* __restrict__ resid,
    float* __restrict__ outF, _Float16* __restrict__ outH) {
    __shared__ __align__(16) _Float16 sA[2][TILE_M * TILE_K];   // [m][k]
    __shared__ __align__(16) _Float16 sB[2][TILE_K * TILE_N];   // [k][n]

    const int tid  = threadIdx.x;
    const int lane = tid & 31;
    const int wid  = tid >> 5;
    const int wm   = wid >> 2;                 // 0..1  (64 rows each)
    const int wn   = wid & 3;                  // 0..3  (32 cols each)
    const int bm   = blockIdx.y * TILE_M;
    const int bn   = blockIdx.x * TILE_N;

    // global staging coordinates (each thread moves 32 bytes of A and of B)
    const int ar  = tid >> 1;                  // 0..127
    const int ac  = (tid & 1) * 16;            // 0 | 16
    const int br  = tid >> 3;                  // 0..31
    const int bc  = (tid & 7) * 16;            // 0..112

    // byte offsets from uniform bases (GVS addressing: SGPR64 base + VGPR i32)
    const unsigned aOff0 = (unsigned)(((size_t)(bm + ar) * K + ac) * 2);
    const unsigned bOff0 = (unsigned)(((size_t)br * Nn + bn + bc) * 2);
    const unsigned long long aBase = (unsigned long long)A;
    const unsigned long long bBase = (unsigned long long)B;

    // prefetch pointers (L2 warm only)
    const _Float16* gA = A + (size_t)(bm + ar) * K + ac;
    const _Float16* gB = B + (size_t)br * Nn + bn + bc;

    // CDNA5 async DMA of one 128x32 A slab + 32x128 B slab into LDS buffer.
    // INST_OFFSET is added to BOTH the LDS and global address, so the two
    // 16-byte chunks per thread share one (vdst, vaddr) pair.
    auto issue = [&](int kt, int buf) {
        unsigned la = (unsigned)(uintptr_t)&sA[buf][ar * TILE_K + ac];
        unsigned lb = (unsigned)(uintptr_t)&sB[buf][br * TILE_N + bc];
        unsigned ga = aOff0 + (unsigned)(kt * TILE_K * 2);
        unsigned gb = bOff0 + (unsigned)(kt * TILE_K * Nn * 2);
        asm volatile(
            "global_load_async_to_lds_b128 %0, %1, %4\n\t"
            "global_load_async_to_lds_b128 %0, %1, %4 offset:16\n\t"
            "global_load_async_to_lds_b128 %2, %3, %5\n\t"
            "global_load_async_to_lds_b128 %2, %3, %5 offset:16"
            :
            : "v"(la), "v"(ga), "v"(lb), "v"(gb), "s"(aBase), "s"(bBase)
            : "memory");
    };

    v8f acc[4][2] = {};

    const int KT = K / TILE_K;
    issue(0, 0);
    for (int kt = 0; kt < KT; ++kt) {
        const int buf = kt & 1;
        if (kt + 1 < KT) {
            issue(kt + 1, buf ^ 1);            // overlap next DMA with compute
            if (kt + 2 < KT) {                 // L2 prefetch two slabs ahead
                __builtin_prefetch(gA + (size_t)(kt + 2) * TILE_K, 0, 0);
                __builtin_prefetch(gB + (size_t)(kt + 2) * TILE_K * Nn, 0, 0);
            }
            // wait until only the 4 just-issued DMAs remain outstanding
            asm volatile("s_wait_asynccnt 0x4" ::: "memory");
        } else {
            asm volatile("s_wait_asynccnt 0x0" ::: "memory");
        }
        __syncthreads();                       // tile kt visible to all waves

        // B fragments: 32x16; lanes 0-15 K=0..15, lanes 16-31 K=16..31
        v16h bf[2];
        {
            const int nc = lane & 15;
            const int kb = (lane >> 4) * 16;
#pragma unroll
            for (int nt = 0; nt < 2; ++nt) {
                const _Float16* p = &sB[buf][kb * TILE_N + wn * 32 + nt * 16 + nc];
#pragma unroll
                for (int i = 0; i < 16; ++i) bf[nt][i] = p[i * TILE_N];
            }
        }
        // A fragments: 16x32; lane<16 rows r K 0..15, lane>=16 K 16..31
        {
            const int r  = lane & 15;
            const int kb = (lane >> 4) * 16;
#pragma unroll
            for (int mt = 0; mt < 4; ++mt) {
                v16h af;
                const _Float16* p = &sA[buf][(wm * 64 + mt * 16 + r) * TILE_K];
#pragma unroll
                for (int i = 0; i < 8; ++i) {
                    af[i]     = p[kb + i];
                    af[8 + i] = p[kb + 8 + i];
                }
#pragma unroll
                for (int nt = 0; nt < 2; ++nt)
                    acc[mt][nt] = __builtin_amdgcn_wmma_f32_16x16x32_f16(
                        false, af, false, bf[nt], (short)0, acc[mt][nt], false, false);
            }
        }
        __syncthreads();   // compute done before next DMA overwrites this buf
    }

    // epilogue: C/D layout -> lane&15 = col, VGPR j -> row j + 8*(lane>=16)
    const int cn   = lane & 15;
    const int rsel = (lane >> 4) * 8;
#pragma unroll
    for (int mt = 0; mt < 4; ++mt) {
#pragma unroll
        for (int nt = 0; nt < 2; ++nt) {
#pragma unroll
            for (int j = 0; j < 8; ++j) {
                const int row = bm + wm * 64 + mt * 16 + rsel + j;
                const int col = bn + wn * 32 + nt * 16 + cn;
                const size_t idx = (size_t)row * Nn + col;
                float v = acc[mt][nt][j];
                if (EPI == 0) {
                    outH[idx] = (_Float16)v;
                } else if (EPI == 1) {
                    outF[idx] = resid[idx] + gamma[col] * (v + bias[col]);
                } else if (EPI == 2) {
                    outH[idx] = (_Float16)gelu_exact(v + bias[col]);
                } else {
                    outF[idx] += gamma[col] * (v + bias[col]);
                }
            }
        }
    }
}

// ---------------------------------------------------------------------------
// L1 column sums over tokens for q and k: inv[(b*2+kind)*512+c] = 1/max(sum,eps)
// ---------------------------------------------------------------------------
__global__ __launch_bounds__(512) void l1sum_kernel(
    const _Float16* __restrict__ qkv, float* __restrict__ inv, int N, int C3) {
    const int bk = blockIdx.x;          // b*2 + kind
    const int b = bk >> 1;
    const int kind = bk & 1;
    const int c = threadIdx.x;          // 0..511
    const _Float16* base = qkv + (size_t)b * N * C3 + kind * 512 + c;
    float s = 0.0f;
    for (int n = 0; n < N; ++n) s += fabsf((float)base[(size_t)n * C3]);
    s = fmaxf(s, 1e-12f);
    inv[(size_t)bk * 512 + c] = 1.0f / s;
}

// ---------------------------------------------------------------------------
// kv' = diag(invq)*diag(invk) * (k^T @ v)  per (b, head), via WMMA. Out f16.
// ---------------------------------------------------------------------------
__global__ __launch_bounds__(256) void kv_wmma_kernel(
    const _Float16* __restrict__ qkv, const float* __restrict__ inv,
    _Float16* __restrict__ kv16, int N, int C3) {
    const int bh = blockIdx.x;          // b*8 + head
    const int b = bh >> 3;
    const int head = bh & 7;
    const int tid = threadIdx.x;
    const int lane = tid & 31;
    const int wid = tid >> 5;
    const int wmd = wid >> 1;           // 0..3 -> 16-row d tile
    const int wne = wid & 1;            // 0..1 -> 32-col e block

    const _Float16* kbase = qkv + (size_t)b * N * C3 + 512 + head * 64;
    const _Float16* vbase = qkv + (size_t)b * N * C3 + 1024 + head * 64;

    v8f acc[2] = {};
    const int r   = lane & 15;
    const int kba = (lane >> 4) * 8;
    const int kbb = (lane >> 4) * 16;
    const int cn  = lane & 15;

    for (int n0 = 0; n0 < N; n0 += 32) {
        v16h af;   // A = k^T : A[r][kk] = k[n0+kk][wmd*16+r]
        const _Float16* kp = kbase + (size_t)n0 * C3 + wmd * 16 + r;
#pragma unroll
        for (int i = 0; i < 8; ++i) {
            af[i]     = kp[(size_t)(kbb + i) * C3];
            af[8 + i] = kp[(size_t)(kbb + 8 + i) * C3];
        }
#pragma unroll
        for (int nt = 0; nt < 2; ++nt) {
            v16h bf;   // B = v : B[kk][e]
            const _Float16* vp = vbase + (size_t)(n0 + kbb) * C3 + wne * 32 + nt * 16 + cn;
#pragma unroll
            for (int i = 0; i < 16; ++i) bf[i] = vp[(size_t)i * C3];
            acc[nt] = __builtin_amdgcn_wmma_f32_16x16x32_f16(
                false, af, false, bf, (short)0, acc[nt], false, false);
        }
    }

    const int rsel = (lane >> 4) * 8;
#pragma unroll
    for (int nt = 0; nt < 2; ++nt) {
#pragma unroll
        for (int j = 0; j < 8; ++j) {
            const int d = wmd * 16 + rsel + j;
            const int e = wne * 32 + nt * 16 + cn;
            const float sc = inv[((size_t)b * 2 + 1) * 512 + head * 64 + d] *
                             inv[((size_t)b * 2 + 0) * 512 + head * 64 + d];
            kv16[((size_t)bh * 64 + d) * 64 + e] = (_Float16)(acc[nt][j] * sc);
        }
    }
}

// ---------------------------------------------------------------------------
// attn = q @ kv'  per (b, head); writes f16 [M, C] (A-matrix for proj GEMM)
// ---------------------------------------------------------------------------
__global__ __launch_bounds__(256) void attn_wmma_kernel(
    const _Float16* __restrict__ qkv, const _Float16* __restrict__ kv16,
    _Float16* __restrict__ outA, int N, int C3, int C) {
    const int bh = blockIdx.x;          // b*8 + head
    const int b = bh >> 3;
    const int head = bh & 7;
    const int n0 = blockIdx.y * 128;
    const int tid = threadIdx.x;
    const int lane = tid & 31;
    const int wmr = tid >> 5;           // 0..7, 16 rows each

    const _Float16* qbase = qkv + (size_t)b * N * C3 + head * 64;

    v8f acc[4] = {};
    const int r   = lane & 15;
    const int kba = (lane >> 4) * 8;
    const int kbb = (lane >> 4) * 16;
    const int cn  = lane & 15;

#pragma unroll
    for (int k0 = 0; k0 < 64; k0 += 32) {
        v16h af;   // A = q rows (contiguous in C3-strided rows)
        const _Float16* qp = qbase + (size_t)(n0 + wmr * 16 + r) * C3 + k0;
#pragma unroll
        for (int i = 0; i < 8; ++i) {
            af[i]     = qp[kbb + i];
            af[8 + i] = qp[kbb + 8 + i];
        }
#pragma unroll
        for (int nt = 0; nt < 4; ++nt) {
            v16h bf;   // B = kv'
            const _Float16* kp = kv16 + ((size_t)bh * 64 + k0 + kbb) * 64 + nt * 16 + cn;
#pragma unroll
            for (int i = 0; i < 16; ++i) bf[i] = kp[i * 64];
            acc[nt] = __builtin_amdgcn_wmma_f32_16x16x32_f16(
                false, af, false, bf, (short)0, acc[nt], false, false);
        }
    }

    const int rsel = (lane >> 4) * 8;
#pragma unroll
    for (int nt = 0; nt < 4; ++nt) {
#pragma unroll
        for (int j = 0; j < 8; ++j) {
            const size_t m = (size_t)b * N + n0 + wmr * 16 + rsel + j;
            const int col = head * 64 + nt * 16 + cn;
            outA[m * C + col] = (_Float16)acc[nt][j];
        }
    }
}

// ---------------------------------------------------------------------------
// depthwise 3x3 conv + GELU + BN(eval)   [B,C,32,32] layout
// ---------------------------------------------------------------------------
__global__ __launch_bounds__(256) void dwconv_bn_kernel(
    const float* __restrict__ yin, const float* __restrict__ w,
    const float* __restrict__ bias, const float* __restrict__ bng,
    const float* __restrict__ bnb, const float* __restrict__ bnm,
    const float* __restrict__ bnv, float* __restrict__ yout) {
    const int idx = blockIdx.x * 256 + threadIdx.x;
    const int wwi = idx & 31;
    const int hh  = (idx >> 5) & 31;
    const int c   = (idx >> 10) & 511;
    const int bi  = idx >> 19;
    const float* plane = yin + ((size_t)bi * 512 + c) * 1024;
    const float* wc = w + c * 9;
    float s = bias[c];
#pragma unroll
    for (int dy = -1; dy <= 1; ++dy) {
        const int hy = hh + dy;
        if (hy < 0 || hy > 31) continue;
#pragma unroll
        for (int dx = -1; dx <= 1; ++dx) {
            const int wx = wwi + dx;
            if (wx < 0 || wx > 31) continue;
            s += wc[(dy + 1) * 3 + (dx + 1)] * plane[hy * 32 + wx];
        }
    }
    s = gelu_exact(s);
    s = (s - bnm[c]) * rsqrtf(bnv[c] + 1e-5f) * bng[c] + bnb[c];
    yout[idx] = s;
}

// depthwise 3x3 conv + bias, accumulate gamma3*val into [B,N,C] residual state
__global__ __launch_bounds__(256) void dwconv_acc_kernel(
    const float* __restrict__ yin, const float* __restrict__ w,
    const float* __restrict__ bias, const float* __restrict__ gamma,
    float* __restrict__ xout) {
    const int idx = blockIdx.x * 256 + threadIdx.x;
    const int wwi = idx & 31;
    const int hh  = (idx >> 5) & 31;
    const int c   = (idx >> 10) & 511;
    const int bi  = idx >> 19;
    const float* plane = yin + ((size_t)bi * 512 + c) * 1024;
    const float* wc = w + c * 9;
    float s = bias[c];
#pragma unroll
    for (int dy = -1; dy <= 1; ++dy) {
        const int hy = hh + dy;
        if (hy < 0 || hy > 31) continue;
#pragma unroll
        for (int dx = -1; dx <= 1; ++dx) {
            const int wx = wwi + dx;
            if (wx < 0 || wx > 31) continue;
            s += wc[(dy + 1) * 3 + (dx + 1)] * plane[hy * 32 + wx];
        }
    }
    const int n = hh * 32 + wwi;
    xout[((size_t)bi * 1024 + n) * 512 + c] += gamma[c] * s;
}

// ---------------------------------------------------------------------------
// host launcher
// ---------------------------------------------------------------------------
extern "C" void kernel_launch(void* const* d_in, const int* in_sizes, int n_in,
                              void* d_out, int out_size, void* d_ws, size_t ws_size,
                              hipStream_t stream) {
    const float* x       = (const float*)d_in[0];
    const float* ln1_g   = (const float*)d_in[3];
    const float* ln1_b   = (const float*)d_in[4];
    const float* qkv_w   = (const float*)d_in[5];
    const float* proj_w  = (const float*)d_in[6];
    const float* proj_b  = (const float*)d_in[7];
    const float* gamma1  = (const float*)d_in[8];
    const float* ln3_g   = (const float*)d_in[9];
    const float* ln3_b   = (const float*)d_in[10];
    const float* conv1_w = (const float*)d_in[11];
    const float* conv1_b = (const float*)d_in[12];
    const float* bn_g    = (const float*)d_in[13];
    const float* bn_b    = (const float*)d_in[14];
    const float* bn_m    = (const float*)d_in[15];
    const float* bn_v    = (const float*)d_in[16];
    const float* conv2_w = (const float*)d_in[17];
    const float* conv2_b = (const float*)d_in[18];
    const float* gamma3  = (const float*)d_in[19];
    const float* ln2_g   = (const float*)d_in[20];
    const float* ln2_b   = (const float*)d_in[21];
    const float* fc1_w   = (const float*)d_in[22];
    const float* fc1_b   = (const float*)d_in[23];
    const float* fc2_w   = (const float*)d_in[24];
    const float* fc2_b   = (const float*)d_in[25];
    const float* gamma2  = (const float*)d_in[26];

    constexpr int Bb = 32, Nt = 1024, Cc = 512, Hd = 2048;
    constexpr int M = Bb * Nt;   // 32768

    char* ws = (char*)d_ws;
    size_t off = 0;
    auto alloc = [&](size_t bytes) -> char* {
        char* p = ws + off;
        off += (bytes + 255) & ~size_t(255);
        return p;
    };
    _Float16* a16     = (_Float16*)alloc((size_t)M * Cc * 2);        //  32 MB
    _Float16* h16     = (_Float16*)alloc((size_t)M * Hd * 2);        // 128 MB
    _Float16* qkv16   = (_Float16*)alloc((size_t)M * 3 * Cc * 2);    //  96 MB
    _Float16* kv16    = (_Float16*)alloc((size_t)Bb * 8 * 64 * 64 * 2);
    float*    invqk   = (float*)alloc((size_t)Bb * 2 * Cc * 4);
    float*    y0      = (float*)alloc((size_t)M * Cc * 4);           //  64 MB
    float*    y1      = (float*)alloc((size_t)M * Cc * 4);           //  64 MB
    _Float16* qkvw16  = (_Float16*)alloc((size_t)Cc * 3 * Cc * 2);
    _Float16* projw16 = (_Float16*)alloc((size_t)Cc * Cc * 2);
    _Float16* fc1w16  = (_Float16*)alloc((size_t)Cc * Hd * 2);
    _Float16* fc2w16  = (_Float16*)alloc((size_t)Hd * Cc * 2);
    float* outF = (float*)d_out;

    // weight conversion
    f32_to_f16_kernel<<<(Cc * 3 * Cc + 255) / 256, 256, 0, stream>>>(qkv_w, qkvw16, Cc * 3 * Cc);
    f32_to_f16_kernel<<<(Cc * Cc + 255) / 256, 256, 0, stream>>>(proj_w, projw16, Cc * Cc);
    f32_to_f16_kernel<<<(Cc * Hd + 255) / 256, 256, 0, stream>>>(fc1_w, fc1w16, Cc * Hd);
    f32_to_f16_kernel<<<(Hd * Cc + 255) / 256, 256, 0, stream>>>(fc2_w, fc2w16, Hd * Cc);

    // --- SimA attention ---
    ln_f16_kernel<<<M, 256, 0, stream>>>(x, ln1_g, ln1_b, a16, Cc);
    gemm_wmma_kernel<0><<<dim3(3 * Cc / TILE_N, M / TILE_M), 256, 0, stream>>>(
        a16, qkvw16, M, Cc, 3 * Cc, nullptr, nullptr, nullptr, nullptr, qkv16);
    l1sum_kernel<<<Bb * 2, 512, 0, stream>>>(qkv16, invqk, Nt, 3 * Cc);
    kv_wmma_kernel<<<Bb * 8, 256, 0, stream>>>(qkv16, invqk, kv16, Nt, 3 * Cc);
    attn_wmma_kernel<<<dim3(Bb * 8, Nt / 128), 256, 0, stream>>>(qkv16, kv16, a16, Nt, 3 * Cc, Cc);
    gemm_wmma_kernel<1><<<dim3(Cc / TILE_N, M / TILE_M), 256, 0, stream>>>(
        a16, projw16, M, Cc, Cc, proj_b, gamma1, x, outF, nullptr);

    // --- LPI (dwconv -> GELU -> BN -> dwconv) ---
    ln_f32_tr_kernel<<<M, 256, 0, stream>>>(outF, ln3_g, ln3_b, y0, Cc, Nt);
    dwconv_bn_kernel<<<(Bb * Cc * Nt) / 256, 256, 0, stream>>>(
        y0, conv1_w, conv1_b, bn_g, bn_b, bn_m, bn_v, y1);
    dwconv_acc_kernel<<<(Bb * Cc * Nt) / 256, 256, 0, stream>>>(
        y1, conv2_w, conv2_b, gamma3, outF);

    // --- MLP ---
    ln_f16_kernel<<<M, 256, 0, stream>>>(outF, ln2_g, ln2_b, a16, Cc);
    gemm_wmma_kernel<2><<<dim3(Hd / TILE_N, M / TILE_M), 256, 0, stream>>>(
        a16, fc1w16, M, Cc, Hd, fc1_b, nullptr, nullptr, nullptr, h16);
    gemm_wmma_kernel<3><<<dim3(Cc / TILE_N, M / TILE_M), 256, 0, stream>>>(
        h16, fc2w16, M, Hd, Cc, fc2_b, gamma2, nullptr, outF, nullptr);
}
